// LearnableCir_10892037063003
// MI455X (gfx1250) — matrix operand order repaired and checked by reading, not compiled, not run on hardware
//
#include <hip/hip_runtime.h>
#include <hip/hip_bf16.h>

typedef __attribute__((ext_vector_type(16))) _Float16 v16h;
typedef __attribute__((ext_vector_type(8)))  _Float16 v8h;
typedef __attribute__((ext_vector_type(8)))  float    v8f;

#define IN_F  256
#define OUT_F 256
#define HW    16
#define NIMG  256
#define ROWS  4                    // output rows per block
#define SROWS (ROWS + 2)           // staged rows incl. halo
#define CSTR  264                  // padded channel stride (halves): 528B = 132 dw -> bank-conflict-free
// wpack: [mt(16)][tap(9)][ks(8)][lane(32)][elem(16)] f16
#define WPACK_ELEMS (16 * 9 * 8 * 32 * 16)   // = 589824 = OUT_F*IN_F*9

// ---------------------------------------------------------------------------
// Kernel 1: effective weight (softmax-blended circulant projections), scattered
// directly into the WMMA A-fragment (16x32 f16) per-lane register layout.
// ---------------------------------------------------------------------------
__global__ void wpack_kernel(const float* __restrict__ W,
                             const float* __restrict__ alphas,
                             _Float16* __restrict__ wpack) {
    int idx = blockIdx.x * blockDim.x + threadIdx.x;
    if (idx >= WPACK_ELEMS) return;

    float mx = alphas[0];
    #pragma unroll
    for (int i = 1; i < 5; ++i) mx = fmaxf(mx, alphas[i]);
    float a[5], s = 0.f;
    #pragma unroll
    for (int i = 0; i < 5; ++i) { a[i] = __expf(alphas[i] - mx); s += a[i]; }
    float inv = 1.f / s;
    #pragma unroll
    for (int i = 0; i < 5; ++i) a[i] *= inv;

    int e     = idx & 15;
    int lane  = (idx >> 4) & 31;
    int ks    = (idx >> 9) & 7;
    int mtTap = idx >> 12;
    int mt    = mtTap / 9;
    int tap   = mtTap % 9;

    int m_  = lane & 15;
    int kG  = lane >> 4;
    int v   = e >> 1;
    int hlf = e & 1;
    int K   = ((v < 4) ? 0 : 16) + 8 * kG + (v & 3) * 2 + hlf;  // A 16x32 f16 layout
    int o   = mt * 16 + m_;
    int ii  = ks * 32 + K;

    const float* Wt = W + tap;

    float val = a[0] * Wt[(o * IN_F + ii) * 9];
    const int bs[4] = {2, 4, 8, 16};
    #pragma unroll
    for (int t = 0; t < 4; ++t) {
        int b  = bs[t];
        int Qb = (o / b) * b;
        int Pb = (ii / b) * b;
        int d  = ((o - ii) % b + b) % b;
        float sum = 0.f;
        for (int j = 0; j < b; ++j) {
            int jj = j + d; if (jj >= b) jj -= b;
            sum += Wt[((Qb + j) * IN_F + (Pb + jj)) * 9];
        }
        val += a[t + 1] * sum / (float)b;
    }
    wpack[idx] = (_Float16)val;
}

// two bank-staggered 16B LDS loads -> one B fragment
__device__ __forceinline__ v16h load_bfrag(const _Float16* p) {
    v8h lo = *(const v8h*)p;
    v8h hi = *(const v8h*)(p + 16);
    v16h r;
    #pragma unroll
    for (int t = 0; t < 8; ++t) { r[t] = lo[t]; r[t + 8] = hi[t]; }
    return r;
}

// ---------------------------------------------------------------------------
// Kernel 2: implicit-GEMM conv via v_wmma_f32_16x16x32_f16.
// One block (8 wave32) per (n, 4-row band). f32->f16 conversion fused into the
// LDS staging. Each A fragment is reused across 4 row accumulators.
// ---------------------------------------------------------------------------
__global__ __launch_bounds__(256) void conv_kernel(const float* __restrict__ xf,
                                                   const _Float16* __restrict__ wpack,
                                                   float* __restrict__ out) {
    __shared__ __attribute__((aligned(32))) _Float16 lds[SROWS * 18 * CSTR]; // 57,024 B

    const int n  = blockIdx.x >> 2;
    const int y0 = (blockIdx.x & 3) * ROWS;

    // ---- stage SROWS halo rows of all 256 channels (f32 -> f16), 1-col zero pad
    {
        const int c = threadIdx.x; // channel
        #pragma unroll
        for (int r = 0; r < SROWS; ++r) {
            int gy = y0 + r - 1;
            _Float16 vals[16];
            if (gy >= 0 && gy < HW) {
                const float4* src = (const float4*)(xf + (((n * IN_F + c) * HW + gy) * HW));
                #pragma unroll
                for (int q = 0; q < 4; ++q) {
                    float4 f = src[q];
                    vals[4 * q + 0] = (_Float16)f.x; vals[4 * q + 1] = (_Float16)f.y;
                    vals[4 * q + 2] = (_Float16)f.z; vals[4 * q + 3] = (_Float16)f.w;
                }
            } else {
                #pragma unroll
                for (int t = 0; t < 16; ++t) vals[t] = (_Float16)0.f;
            }
            lds[(r * 18 + 0)  * CSTR + c] = (_Float16)0.f;
            lds[(r * 18 + 17) * CSTR + c] = (_Float16)0.f;
            #pragma unroll
            for (int t = 0; t < 16; ++t)
                lds[(r * 18 + (t + 1)) * CSTR + c] = vals[t];
        }
    }
    __syncthreads();

    const int lane = threadIdx.x & 31;
    const int wv   = threadIdx.x >> 5;
    const int kG   = lane >> 4;
    const int colW = lane & 15;

    #pragma unroll
    for (int t2 = 0; t2 < 2; ++t2) {
        const int mt = wv + 8 * t2;
        v8f acc0 = {}, acc1 = {}, acc2 = {}, acc3 = {};
        const _Float16* wbase = wpack + mt * (9 * 8 * 32 * 16);
        #pragma unroll 1
        for (int tap = 0; tap < 9; ++tap) {
            const int kh = tap / 3;
            const int kw = tap - 3 * kh;
            const _Float16* abase = wbase + tap * (8 * 32 * 16);
            __builtin_prefetch((const void*)(abase + (8 * 32 * 16) + lane * 16), 0, 1);
            const int lb = (kh * 18 + colW + kw) * CSTR + kG * 8;
            #pragma unroll
            for (int ks = 0; ks < 8; ++ks) {
                v16h afrag = *(const v16h*)(abase + (ks * 32 + lane) * 16);
                v16h b0 = load_bfrag(&lds[lb + (0 * 18) * CSTR + ks * 32]);
                acc0 = __builtin_amdgcn_wmma_f32_16x16x32_f16(false, afrag, false, b0,
                                                              (short)0, acc0, false, false);
                v16h b1 = load_bfrag(&lds[lb + (1 * 18) * CSTR + ks * 32]);
                acc1 = __builtin_amdgcn_wmma_f32_16x16x32_f16(false, afrag, false, b1,
                                                              (short)0, acc1, false, false);
                v16h b2 = load_bfrag(&lds[lb + (2 * 18) * CSTR + ks * 32]);
                acc2 = __builtin_amdgcn_wmma_f32_16x16x32_f16(false, afrag, false, b2,
                                                              (short)0, acc2, false, false);
                v16h b3 = load_bfrag(&lds[lb + (3 * 18) * CSTR + ks * 32]);
                acc3 = __builtin_amdgcn_wmma_f32_16x16x32_f16(false, afrag, false, b3,
                                                              (short)0, acc3, false, false);
            }
        }
        // D layout: lane half kG, VGPR v -> out channel mt*16 + v + 8*kG, col = colW
        float* ob = out + (((n * OUT_F + mt * 16 + 8 * kG) * HW + y0) * HW + colW);
        #pragma unroll
        for (int v = 0; v < 8; ++v) ob[v * (HW * HW) + 0 * HW] = acc0[v];
        #pragma unroll
        for (int v = 0; v < 8; ++v) ob[v * (HW * HW) + 1 * HW] = acc1[v];
        #pragma unroll
        for (int v = 0; v < 8; ++v) ob[v * (HW * HW) + 2 * HW] = acc2[v];
        #pragma unroll
        for (int v = 0; v < 8; ++v) ob[v * (HW * HW) + 3 * HW] = acc3[v];
    }
}

// ---------------------------------------------------------------------------
extern "C" void kernel_launch(void* const* d_in, const int* in_sizes, int n_in,
                              void* d_out, int out_size, void* d_ws, size_t ws_size,
                              hipStream_t stream) {
    const float* x      = (const float*)d_in[0];
    const float* weight = (const float*)d_in[1];
    const float* alphas = (const float*)d_in[2];
    float* out          = (float*)d_out;

    _Float16* wpack = (_Float16*)d_ws;

    wpack_kernel<<<(WPACK_ELEMS + 255) / 256, 256, 0, stream>>>(weight, alphas, wpack);
    conv_kernel<<<NIMG * (HW / ROWS), 256, 0, stream>>>(x, wpack, out);
}
